// Correlation_40097814676030
// MI455X (gfx1250) — compile-verified, matrix operand order checked by reading, not run on hardware
//
#include <hip/hip_runtime.h>

typedef __attribute__((ext_vector_type(2)))  __bf16 v2bf;
typedef __attribute__((ext_vector_type(8)))  __bf16 v8bf;
typedef __attribute__((ext_vector_type(16))) __bf16 v16bf;
typedef __attribute__((ext_vector_type(8)))  float  v8f;

namespace {
constexpr int kB = 8, kC = 256, kH = 96, kW = 128;
constexpr int kD = 21;                 // displacements per axis
constexpr int kQ = kD * kD;            // 441 output channels
constexpr int PITCH   = 264;           // bf16 elems per LDS row (256 + 8 pad -> conflict-free b128 frag reads)
constexpr int A_ROWS  = 128;           // 2 parities * 64 pixels
constexpr int B_ROWS  = 192;           // 2 parities * (16 guard + 64 + 16 guard)
constexpr int A_ELEMS = A_ROWS * PITCH;
constexpr int B_ELEMS = B_ROWS * PITCH;
constexpr size_t SMEM_BYTES = (size_t)(A_ELEMS + 2 * B_ELEMS) * sizeof(unsigned short); // 270,336 B < 320 KB
}

// channels (c, c+1) -> one packed bf16x2 dword; lets the backend use the
// native f32->bf16 convert (v_cvt_pk_bf16_f32-class) instead of integer RNE emulation
static __device__ __forceinline__ v2bf cvt2bf(float a, float b) {
  v2bf r;
  r[0] = (__bf16)a;
  r[1] = (__bf16)b;
  return r;
}

// 16x32 bf16 WMMA fragment for one lane: lanes 0-15 hold K = {0..7, 16..23},
// lanes 16-31 hold K = {8..15, 24..31} (caller applies the +8 channel offset).
// Two aligned 16B LDS reads -> ds_load_b128.
static __device__ __forceinline__ v16bf load_frag(const unsigned short* p) {
  v8bf lo = *reinterpret_cast<const v8bf*>(p);
  v8bf hi = *reinterpret_cast<const v8bf*>(p + 16);
  return __builtin_shufflevector(lo, hi, 0,1,2,3,4,5,6,7,8,9,10,11,12,13,14,15);
}

__global__ __launch_bounds__(256) void corr441_wmma_bf16(
    const float* __restrict__ in1, const float* __restrict__ in2,
    float* __restrict__ out) {
  extern __shared__ unsigned short smem[];
  unsigned short* As = smem;                       // A: [parity][64 pixels][PITCH]
  // B buffers: [parity][96 slots][PITCH], slots = 16 guard + 64 pixels + 16 guard

  const int t    = threadIdx.x;
  const int lane = t & 31;       // wave32
  const int wv   = t >> 5;       // 8 waves
  const int y    = blockIdx.x;
  const int b    = blockIdx.y;

  const int par  = wv >> 2;          // parity owned by this wave
  const int m0   = (wv & 3) << 4;    // 16-pixel row-tile base (within 64)
  const int lo16 = lane & 15;
  const int hi   = lane >> 4;
  const int koff = hi << 3;          // +8 channel sub-offset for upper half-wave

  // ---- vectorized staging: float4 along x, 2 channels -> packed bf16x2 dword ----
  // thread t: x-quad s = t&31 (x = 4s..4s+3), channel pair base = 2*(t>>5), step 16
  auto stageRow = [&](const float* rowBase, bool valid, unsigned short* dstPar0,
                      int rstride, int guard) {
    const int s  = t & 31;
    const int x4 = s << 2;
    unsigned short* d00 = dstPar0 + (guard + 2 * s) * PITCH;            // p=0, m=2s
    unsigned short* d10 = dstPar0 + (rstride + guard + 2 * s) * PITCH;  // p=1, m=2s
    unsigned short* d01 = d00 + PITCH;                                  // p=0, m=2s+1
    unsigned short* d11 = d10 + PITCH;                                  // p=1, m=2s+1
    for (int c = (t >> 5) << 1; c < kC; c += 16) {
      float4 fa, fb;
      if (valid) {
        fa = *reinterpret_cast<const float4*>(rowBase + (size_t)c * (kH * kW) + x4);
        fb = *reinterpret_cast<const float4*>(rowBase + (size_t)(c + 1) * (kH * kW) + x4);
      } else {
        fa = make_float4(0.f, 0.f, 0.f, 0.f);
        fb = fa;
      }
      *reinterpret_cast<v2bf*>(d00 + c) = cvt2bf(fa.x, fb.x);
      *reinterpret_cast<v2bf*>(d10 + c) = cvt2bf(fa.y, fb.y);
      *reinterpret_cast<v2bf*>(d01 + c) = cvt2bf(fa.z, fb.z);
      *reinterpret_cast<v2bf*>(d11 + c) = cvt2bf(fa.w, fb.w);
    }
  };

  auto stageB = [&](int dyi, unsigned short* Bbuf) {
    const int  ys    = y + 2 * dyi - 20;
    const bool valid = (ys >= 0) && (ys < kH);
    const int  ysc   = valid ? ys : 0;
    stageRow(in2 + ((size_t)b * kC * kH + ysc) * kW, valid, Bbuf, 96, 16);
  };

  auto prefetchRow = [&](int ysn) {                // one channel per thread, 4 lines
    if (ysn >= 0 && ysn < kH) {
      const float* rp = in2 + (((size_t)b * kC + t) * kH + ysn) * kW;
      #pragma unroll
      for (int s = 0; s < 4; ++s) __builtin_prefetch(rp + s * 32, 0, 1);
    }
  };

  // ---- stage in1 row y (once) ----
  stageRow(in1 + ((size_t)b * kC * kH + y) * kW, true, As, 64, 0);

  // ---- zero the guard slots of both B buffers (once) ----
  for (int idx = t; idx < 2 * 64 * PITCH; idx += 256) {
    int buf  = idx / (64 * PITCH);
    int r    = idx - buf * (64 * PITCH);
    int row  = r / PITCH;
    int c    = r - row * PITCH;
    int pp   = row >> 5;
    int s    = row & 31;
    int slot = (s < 16) ? s : (64 + s);            // slots [0,16) and [80,96)
    smem[A_ELEMS + buf * B_ELEMS + (pp * 96 + slot) * PITCH + c] = 0;
  }

  stageB(0, smem + A_ELEMS);
  __syncthreads();

  // ---- hoist the 8 A fragments (K = 256 as 8 x K32) for all 21 dy steps ----
  v16bf afrag[8];
  {
    const unsigned short* ap = As + (par * 64 + m0 + lo16) * PITCH + koff;
    #pragma unroll
    for (int k = 0; k < 8; ++k) afrag[k] = load_frag(ap + 32 * k);
  }

  for (int dyi = 0; dyi < kD; ++dyi) {
    const unsigned short* Bb = smem + A_ELEMS + (size_t)(dyi & 1) * B_ELEMS;
    prefetchRow(y + 2 * (dyi + 1) - 20);

    #pragma unroll
    for (int j = 0; j < 3; ++j) {                  // 3 column tiles cover |n-m| <= 10
      v8f acc = {};
      const unsigned short* bp =
          Bb + (par * 96 + m0 + 16 * j + lo16) * PITCH + koff; // slot = 16 + (m0-16+16j) + lo16
      #pragma unroll
      for (int k = 0; k < 8; ++k) {
        v16bf bfrag = load_frag(bp + 32 * k);
        acc = __builtin_amdgcn_wmma_f32_16x16x32_bf16(
            false, afrag[k], false, bfrag, (short)0, acc, false, false);
      }
      // D layout: lane -> N, VGPR v -> M = v + 8*(lane>=16). Band scatter.
      const int dbase = 16 * j - 16 + lo16 - 8 * hi;   // diff = dbase - v
      #pragma unroll
      for (int v = 0; v < 8; ++v) {
        const int diff = dbase - v;
        if ((unsigned)(diff + 10) <= 20u) {
          const int q  = dyi * 21 + diff + 10;
          const int mg = m0 + v + 8 * hi;
          out[(((b * kQ) + q) * kH + y) * kW + (2 * mg + par)] = acc[v];
        }
      }
    }

    if (dyi + 1 < kD)
      stageB(dyi + 1, smem + A_ELEMS + (size_t)((dyi + 1) & 1) * B_ELEMS);
    __syncthreads();
  }
}

extern "C" void kernel_launch(void* const* d_in, const int* in_sizes, int n_in,
                              void* d_out, int out_size, void* d_ws, size_t ws_size,
                              hipStream_t stream) {
  const float* in1 = (const float*)d_in[0];
  const float* in2 = (const float*)d_in[1];
  float* out = (float*)d_out;
  (void)in_sizes; (void)n_in; (void)d_ws; (void)ws_size; (void)out_size;

  dim3 grid(kH, kB, 1);   // one workgroup per (y, b); 8 waves = 2 parities x 4 row tiles
  corr441_wmma_bf16<<<grid, dim3(256, 1, 1), SMEM_BYTES, stream>>>(in1, in2, out);
}
